// LatentMemoryModule_25005299598046
// MI455X (gfx1250) — matrix operand order repaired and checked by reading, not compiled, not run on hardware
//
#include <hip/hip_runtime.h>
#include <math.h>

#define NB   4
#define TT   4096
#define NT   16384      // B*T tokens
#define HD   2048
#define DD   128
#define LSEG 64
#define NSEG 64
#define NCOL 256        // [sp_w(128) | W_q(128)]
#define SCALE_D 0.08838834764831845f  // 1/sqrt(128)

typedef __bf16 v16bf __attribute__((ext_vector_type(16)));
typedef float  v8f   __attribute__((ext_vector_type(8)));
typedef int    v4i   __attribute__((ext_vector_type(4)));

union BFrag { v16bf v; unsigned short u[16]; };

__device__ __forceinline__ unsigned short f2bf(float f) {
  union { float f; unsigned int u; } x; x.f = f;
  unsigned int u = x.u;
  return (unsigned short)((u + 0x7FFFu + ((u >> 16) & 1u)) >> 16);
}

// ---- CDNA5 async global->LDS copy (ASYNCcnt path), guarded with fallback ----
#if defined(__AMDGCN__) && __has_builtin(__builtin_amdgcn_global_load_async_to_lds_b128)
#define USE_ASYNC 1
typedef __attribute__((address_space(1))) v4i* gas_v4i_ptr;
typedef __attribute__((address_space(3))) v4i* las_v4i_ptr;
__device__ __forceinline__ void async_cp16(const void* gsrc, void* ldst) {
  __builtin_amdgcn_global_load_async_to_lds_b128(
      (gas_v4i_ptr)(gsrc), (las_v4i_ptr)(ldst), 0, 0);
}
#if __has_builtin(__builtin_amdgcn_s_wait_asynccnt)
#define WAIT_ASYNC() __builtin_amdgcn_s_wait_asynccnt(0)
#else
#define WAIT_ASYNC() asm volatile("s_wait_asynccnt 0" ::: "memory")
#endif
#else
#define USE_ASYNC 0
#endif

// ---------------- prep: sgw = sum(g*W_eta), sbw = sum(b*W_eta); zero loss accumulators
__global__ void lmm_prep(const float* g, const float* b, const float* w, float* scal) {
  __shared__ float r1[256], r2[256];
  float s1 = 0.f, s2 = 0.f;
  for (int h = threadIdx.x; h < HD; h += 256) { s1 += g[h] * w[h]; s2 += b[h] * w[h]; }
  r1[threadIdx.x] = s1; r2[threadIdx.x] = s2; __syncthreads();
  for (int st = 128; st > 0; st >>= 1) {
    if (threadIdx.x < st) { r1[threadIdx.x] += r1[threadIdx.x + st]; r2[threadIdx.x] += r2[threadIdx.x + st]; }
    __syncthreads();
  }
  if (threadIdx.x == 0) { scal[0] = r1[0]; scal[1] = r2[0]; scal[2] = 0.f; scal[3] = 0.f; }
}

// ---------------- convert [sp_w | W_q] -> bf16 Wcat[2048][256]
__global__ void lmm_wcat(const float* spw, const float* wqkv, unsigned short* Wcat) {
  int idx = blockIdx.x * 256 + threadIdx.x;
  int h = idx >> 8, n = idx & 255;
  float v = (n < DD) ? spw[h * DD + n] : wqkv[h * (3 * DD) + 2 * DD + (n - DD)];
  Wcat[idx] = f2bf(v);
}

// ---------------- per-token LN stats + eta partial dot (one wave32 per token)
__global__ void lmm_lnstats(const float* H, const float* g, const float* w,
                            float* mu, float* rstd, float* s3o) {
  int wave = threadIdx.x >> 5, lane = threadIdx.x & 31;
  int tok = blockIdx.x * 8 + wave;
  const float* hp = H + (long)tok * HD;
  float s1 = 0.f, s2 = 0.f, s3 = 0.f;
  for (int h = lane; h < HD; h += 32) {
    float x = hp[h];
    s1 += x; s2 += x * x; s3 += x * g[h] * w[h];
  }
  for (int m = 16; m; m >>= 1) {
    s1 += __shfl_xor(s1, m, 32);
    s2 += __shfl_xor(s2, m, 32);
    s3 += __shfl_xor(s3, m, 32);
  }
  if (lane == 0) {
    float m1 = s1 * (1.f / HD);
    float var = s2 * (1.f / HD) - m1 * m1;
    mu[tok] = m1; rstd[tok] = rsqrtf(var + 1e-5f); s3o[tok] = s3;
  }
}

// ---------------- fused LN + dual projection GEMM: P[tok][0..127]=H_mem(+bias), [128..255]=Q_r
// 64x256 output tile per block, K=2048 streamed in 64-chunks, bf16 WMMA.
// Also emits Q_r as bf16 (Qbf) straight from accumulators for the read kernel.
__global__ void lmm_gemm(const float* H, const float* mu, const float* rstd,
                         const float* g, const float* bia, const unsigned short* Wcat,
                         const float* spb, float* P, unsigned short* Qbf) {
  __shared__ unsigned short As[64 * 72];
  __shared__ unsigned short Bs[64 * 264];
  int mblk = blockIdx.x;
  int tid = threadIdx.x;
  int wave = tid >> 5, lane = tid & 31;
  int tm = wave & 3;
  int tnb = (wave >> 2) * 8;
  v8f acc[8] = {};
  for (int k0 = 0; k0 < HD; k0 += 64) {
    // A tile: 64 tokens x 64 k, LN applied on the fly (float4 loads, packed bf16 stores)
    for (int i = 0; i < 4; i++) {
      int idx = tid + i * 256;          // 0..1023
      int r = idx >> 4;                 // token row 0..63
      int c4 = (idx & 15) * 4;          // k col 0..60
      int tok = mblk * 64 + r;
      int h = k0 + c4;
      float4 hv = *(const float4*)(H + (long)tok * HD + h);
      float4 gv = *(const float4*)(g + h);
      float4 bv = *(const float4*)(bia + h);
      float m1 = mu[tok], rs = rstd[tok];
      unsigned int lo = (unsigned int)f2bf((hv.x - m1) * rs * gv.x + bv.x)
                      | ((unsigned int)f2bf((hv.y - m1) * rs * gv.y + bv.y) << 16);
      unsigned int hi = (unsigned int)f2bf((hv.z - m1) * rs * gv.z + bv.z)
                      | ((unsigned int)f2bf((hv.w - m1) * rs * gv.w + bv.w) << 16);
      unsigned int* dst = (unsigned int*)(As + r * 72 + c4);
      dst[0] = lo; dst[1] = hi;
    }
    // B tile: 64 k x 256 n (pure copy -> async to LDS when available)
#if USE_ASYNC
    for (int i = 0; i < 8; i++) {
      int idx = tid + i * 256;          // 0..2047 16B chunks
      int r = idx >> 5;
      int c8 = (idx & 31) * 8;
      async_cp16(Wcat + (k0 + r) * NCOL + c8, Bs + r * 264 + c8);
    }
    WAIT_ASYNC();
#else
    for (int i = 0; i < 32; i++) {
      int idx = tid + i * 256;          // 0..8191 dwords
      int r = idx >> 7;
      int c2 = (idx & 127) * 2;
      *(unsigned int*)(Bs + r * 264 + c2) =
          *(const unsigned int*)(Wcat + (k0 + r) * NCOL + c2);
    }
#endif
    __syncthreads();
    for (int ks = 0; ks < 64; ks += 32) {
      BFrag a;
      int ml = tm * 16 + (lane & 15);
      int kb = ks + ((lane >> 4) << 3);
#pragma unroll
      for (int e = 0; e < 16; e++) a.u[e] = As[ml * 72 + kb + ((e >> 3) << 4) + (e & 7)];
      int kr = ks + lane;
#pragma unroll
      for (int tt = 0; tt < 8; tt++) {
        BFrag bb;
        int n0 = (tnb + tt) * 16;
#pragma unroll
        for (int e = 0; e < 16; e++) bb.u[e] = Bs[(kb + ((e >> 3) << 4) + (e & 7)) * 264 + n0 + (lane & 15)];
        acc[tt] = __builtin_amdgcn_wmma_f32_16x16x32_bf16(false, a.v, false, bb.v,
                                                          (short)0, acc[tt], false, false);
      }
    }
    __syncthreads();
  }
  int mo = 8 * (lane >> 4), nlo = lane & 15;
#pragma unroll
  for (int tt = 0; tt < 8; tt++) {
    int n = (tnb + tt) * 16 + nlo;
    float bias = (n < DD) ? spb[n] : 0.f;
#pragma unroll
    for (int r = 0; r < 8; r++) {
      int m = mblk * 64 + tm * 16 + r + mo;
      float v = acc[tt][r] + bias;
      P[(long)m * NCOL + n] = v;
      if (n >= DD) Qbf[(long)m * DD + (n - DD)] = f2bf(v);
    }
  }
}

// ---------------- per-segment summary: Z_seg + eta_seg (one block per (b,s))
__global__ void lmm_segment(const float* P, const float* sumq, const float* mu,
                            const float* rstd, const float* s3, const float* scal,
                            const float* wetab, float* Zseg, float* etaseg) {
  __shared__ float sc[LSEG], sw[LSEG], se[LSEG];
  int bx = blockIdx.x;          // bx = b*64 + s
  int gbase = bx * LSEG;        // = b*4096 + s*64
  int t = threadIdx.x;
  if (t < LSEG) {
    int gtok = gbase + t;
    const float* p = P + (long)gtok * NCOL;
    float acc = 0.f;
    for (int d = 0; d < DD; d++) acc += sumq[d] * p[d];
    sc[t] = acc * SCALE_D;
    float e = rstd[gtok] * (s3[gtok] - mu[gtok] * scal[0]) + scal[1] + wetab[0];
    se[t] = 1.f / (1.f + expf(-e));
  }
  __syncthreads();
  if (t == 0) {
    float mx = sc[0], em = se[0];
    for (int l = 1; l < LSEG; l++) { mx = fmaxf(mx, sc[l]); em = fmaxf(em, se[l]); }
    float s = 0.f;
    for (int l = 0; l < LSEG; l++) { float e = expf(sc[l] - mx); sw[l] = e; s += e; }
    float inv = 1.f / s;
    for (int l = 0; l < LSEG; l++) sw[l] *= inv;
    etaseg[bx] = em;
  }
  __syncthreads();
  int d = t;  // 128 threads, one per D channel
  float z = 0.f;
  for (int l = 0; l < LSEG; l++) z += sw[l] * P[(long)(gbase + l) * NCOL + d];
  Zseg[(long)bx * DD + d] = z;
}

// ---------------- memory update + loss terms (one block per batch)
// Emits M_new in f32 (for loss), bf16 row-major, and bf16 transposed (for read kernel).
__global__ void lmm_update(const float* Zin, const float* etaseg, const float* M,
                           const float* etach, const float* temp, float* Mnew,
                           unsigned short* Mbf, unsigned short* Mtbf, float* scal) {
  extern __shared__ float uds[];
  float* Zs  = uds;         // 64*128
  float* Asf = uds + 8192;  // 64*128
  __shared__ float red[256];
  __shared__ float nrr[DD];
  __shared__ float sEtaAvg;
  int b = blockIdx.x, tid = threadIdx.x;
  const float* Mb = M + (long)b * DD * DD;
  float* Mo = Mnew + (long)b * DD * DD;
  unsigned short* Mob  = Mbf  + (long)b * DD * DD;
  unsigned short* Mot  = Mtbf + (long)b * DD * DD;
  for (int i = 0; i < 32; i++) Zs[tid + i * 256] = Zin[(long)b * 8192 + tid + i * 256];
  if (tid == 0) {
    float s = 0.f;
    for (int l = 0; l < NSEG; l++) s += etaseg[b * NSEG + l];
    sEtaAvg = s * (1.f / NSEG);
  }
  __syncthreads();
  float texp = SCALE_D * expf(temp[0]);
  for (int i = 0; i < 32; i++) {
    int idx = tid + i * 256;
    int s = idx >> 7, k = idx & 127;
    const float* mk = Mb + k * DD;
    const float* zz = Zs + s * DD;
    float acc = 0.f;
    for (int d = 0; d < DD; d++) acc += zz[d] * mk[d];
    Asf[idx] = acc * texp;
  }
  __syncthreads();
  if (tid < NSEG) {
    float* row = Asf + tid * DD;
    float mx = row[0];
    for (int k = 1; k < DD; k++) mx = fmaxf(mx, row[k]);
    float s = 0.f;
    for (int k = 0; k < DD; k++) { float e = expf(row[k] - mx); row[k] = e; s += e; }
    float inv = 1.f / s;
    for (int k = 0; k < DD; k++) row[k] *= inv;
  }
  __syncthreads();
  for (int i = 0; i < 32; i++) {
    int idx = tid + i * 256;
    Zs[idx] *= etaseg[b * NSEG + (idx >> 7)];
  }
  __syncthreads();
  float nacc = 0.f;
  for (int i = 0; i < 64; i++) {
    int idx = tid + i * 256;
    int k = idx >> 7, d = idx & 127;
    float delta = 0.f;
    for (int s = 0; s < NSEG; s++) delta += Asf[s * DD + k] * Zs[s * DD + d];
    float gate = sEtaAvg * (1.f / (1.f + expf(-etach[d])));
    float mn = (1.f - gate) * Mb[idx] + delta * (1.f / NSEG);
    Mo[idx] = mn;
    unsigned short mh = f2bf(mn);
    Mob[idx] = mh;
    Mot[d * DD + k] = mh;
    nacc += delta * delta;
  }
  red[tid] = nacc; __syncthreads();
  for (int st = 128; st > 0; st >>= 1) { if (tid < st) red[tid] += red[tid + st]; __syncthreads(); }
  if (tid == 0) atomicAdd(&scal[2], red[0]);
  __threadfence();
  __syncthreads();
  if (tid < DD) {
    const float* r = Mo + tid * DD;
    float s = 0.f;
    for (int d = 0; d < DD; d++) s += r[d] * r[d];
    nrr[tid] = 1.f / fmaxf(sqrtf(s), 1e-12f);
  }
  __syncthreads();
  float dacc = 0.f;
  for (int i = 0; i < 64; i++) {
    int idx = tid + i * 256;
    int k = idx >> 7, j = idx & 127;
    if (k != j) {
      const float* rk = Mo + k * DD;
      const float* rj = Mo + j * DD;
      float s = 0.f;
      for (int d = 0; d < DD; d++) s += rk[d] * rj[d];
      float sim = s * nrr[k] * nrr[j];
      dacc += sim * sim;
    }
  }
  red[tid] = dacc; __syncthreads();
  for (int st = 128; st > 0; st >>= 1) { if (tid < st) red[tid] += red[tid + st]; __syncthreads(); }
  if (tid == 0) atomicAdd(&scal[3], red[0]);
}

__global__ void lmm_finalize(const float* scal, float* out) {
  out[(long)NT * DD] = scal[2] * (1.f / NSEG) * 0.01f
                     + scal[3] * (1.f / (NB * DD * DD)) * 0.1f;
}

// ---------------- read attention: scores WMMA -> softmax -> output WMMA (64 tokens/block)
// All tile stagings are pure bf16 copies -> async global->LDS when available.
__global__ void lmm_read(const unsigned short* Qbf, const unsigned short* Mbf,
                         const unsigned short* Mtbf, float* out) {
  extern __shared__ char rds[];
  unsigned short* Mb = (unsigned short*)rds;     // 128*128 bf16 (32KB) row-major [k][d]
  unsigned short* Mt = Mb + DD * DD;             // 128*128 bf16 (32KB) transposed [d][k]
  unsigned short* Qb = Mt + DD * DD;             // 64*128 bf16 (16KB)
  float* Sc = (float*)(Qb + LSEG * DD);          // 64*128 f32 (32KB)
  int blk = blockIdx.x;
  int b = blk >> 6;
  int g0 = blk * LSEG;                           // global token base
  int tid = threadIdx.x, wave = tid >> 5, lane = tid & 31;
  const unsigned short* Mgb = Mbf  + (long)b * DD * DD;
  const unsigned short* Mgt = Mtbf + (long)b * DD * DD;
  const unsigned short* Qg  = Qbf  + (long)g0 * DD;
#if USE_ASYNC
  for (int i = 0; i < 8; i++) { int idx = tid + i * 256; async_cp16(Mgb + idx * 8, Mb + idx * 8); }
  for (int i = 0; i < 8; i++) { int idx = tid + i * 256; async_cp16(Mgt + idx * 8, Mt + idx * 8); }
  for (int i = 0; i < 4; i++) { int idx = tid + i * 256; async_cp16(Qg + idx * 8, Qb + idx * 8); }
  WAIT_ASYNC();
#else
  for (int i = 0; i < 32; i++) {
    int idx = tid + i * 256;
    *(unsigned int*)(Mb + idx * 2) = *(const unsigned int*)(Mgb + idx * 2);
    *(unsigned int*)(Mt + idx * 2) = *(const unsigned int*)(Mgt + idx * 2);
  }
  for (int i = 0; i < 16; i++) {
    int idx = tid + i * 256;
    *(unsigned int*)(Qb + idx * 2) = *(const unsigned int*)(Qg + idx * 2);
  }
#endif
  __syncthreads();
  int tm = wave & 3;
  int tnb = (wave >> 2) * 4;
  int mo = 8 * (lane >> 4), nlo = lane & 15;
  {
    v8f acc[4] = {};
    for (int ks = 0; ks < DD; ks += 32) {
      BFrag a;
      int ml = tm * 16 + (lane & 15);
      int kb = ks + ((lane >> 4) << 3);
#pragma unroll
      for (int e = 0; e < 16; e++) a.u[e] = Qb[ml * DD + kb + ((e >> 3) << 4) + (e & 7)];
      int kr = ks + lane;
#pragma unroll
      for (int tt = 0; tt < 4; tt++) {
        int tn = tnb + tt;
        BFrag bb;                                   // B = M^T from Mt (contiguous)
#pragma unroll
        for (int e = 0; e < 16; e++) bb.u[e] = Mt[(kb + ((e >> 3) << 4) + (e & 7)) * DD + tn * 16 + (lane & 15)];
        acc[tt] = __builtin_amdgcn_wmma_f32_16x16x32_bf16(false, a.v, false, bb.v,
                                                          (short)0, acc[tt], false, false);
      }
    }
#pragma unroll
    for (int tt = 0; tt < 4; tt++) {
      int tn = tnb + tt;
#pragma unroll
      for (int r = 0; r < 8; r++)
        Sc[(tm * 16 + r + mo) * DD + tn * 16 + nlo] = acc[tt][r];
    }
  }
  __syncthreads();
  if (tid < LSEG) {                                 // softmax(2*x) over K=128
    float* row = Sc + tid * DD;
    float mx = row[0];
    for (int k = 1; k < DD; k++) mx = fmaxf(mx, row[k]);
    float s = 0.f;
    for (int k = 0; k < DD; k++) { float e = expf(2.f * (row[k] - mx)); row[k] = e; s += e; }
    float inv = 1.f / s;
    unsigned short* q = Qb + tid * DD;              // reuse Qb for A_r (bf16)
    for (int k = 0; k < DD; k++) q[k] = f2bf(row[k] * inv);
  }
  __syncthreads();
  {
    v8f acc[4] = {};
    for (int ks = 0; ks < DD; ks += 32) {
      BFrag a;
      int ml = tm * 16 + (lane & 15);
      int kb = ks + ((lane >> 4) << 3);
#pragma unroll
      for (int e = 0; e < 16; e++) a.u[e] = Qb[ml * DD + kb + ((e >> 3) << 4) + (e & 7)];
      int kr = ks + lane;
#pragma unroll
      for (int tt = 0; tt < 4; tt++) {
        int tn = tnb + tt;
        BFrag bb;                                   // B = M from Mb (contiguous)
#pragma unroll
        for (int e = 0; e < 16; e++) bb.u[e] = Mb[(kb + ((e >> 3) << 4) + (e & 7)) * DD + tn * 16 + (lane & 15)];
        acc[tt] = __builtin_amdgcn_wmma_f32_16x16x32_bf16(false, a.v, false, bb.v,
                                                          (short)0, acc[tt], false, false);
      }
    }
#pragma unroll
    for (int tt = 0; tt < 4; tt++) {
      int tn = tnb + tt;
#pragma unroll
      for (int r = 0; r < 8; r++) {
        int m = tm * 16 + r + mo;
        out[(long)(g0 + m) * DD + tn * 16 + nlo] = acc[tt][r];
      }
    }
  }
}

extern "C" void kernel_launch(void* const* d_in, const int* in_sizes, int n_in,
                              void* d_out, int out_size, void* d_ws, size_t ws_size,
                              hipStream_t stream) {
  (void)in_sizes; (void)n_in; (void)out_size; (void)ws_size;
  const float* H     = (const float*)d_in[0];
  const float* M     = (const float*)d_in[1];
  const float* ln_g  = (const float*)d_in[2];
  const float* ln_b  = (const float*)d_in[3];
  const float* Wew   = (const float*)d_in[4];
  const float* Web   = (const float*)d_in[5];
  const float* sumq  = (const float*)d_in[6];
  const float* spw   = (const float*)d_in[7];
  const float* spb   = (const float*)d_in[8];
  const float* etach = (const float*)d_in[9];
  const float* temp  = (const float*)d_in[10];
  const float* wqkv  = (const float*)d_in[11];
  float* out = (float*)d_out;

  float* mu   = (float*)d_ws;
  float* rstd = mu + NT;
  float* s3   = rstd + NT;
  float* scal = s3 + NT;                                   // [sgw, sbw, norm_acc, div_acc]
  unsigned short* Wcat = (unsigned short*)(scal + 4);      // 2048*256 bf16
  float* P    = (float*)(Wcat + (size_t)HD * NCOL);        // 16384*256 f32
  float* Zseg = P + (size_t)NT * NCOL;                     // 4*64*128
  float* etas = Zseg + (size_t)NB * NSEG * DD;             // 256
  float* Mnew = etas + NB * NSEG;                          // 4*128*128 f32
  unsigned short* Qbf  = (unsigned short*)(Mnew + (size_t)NB * DD * DD); // 16384*128 bf16
  unsigned short* Mbf  = Qbf + (size_t)NT * DD;            // 4*128*128 bf16
  unsigned short* Mtbf = Mbf + (size_t)NB * DD * DD;       // 4*128*128 bf16 (transposed)

  (void)hipFuncSetAttribute(reinterpret_cast<const void*>(lmm_update),
                            hipFuncAttributeMaxDynamicSharedMemorySize, 65536);
  (void)hipFuncSetAttribute(reinterpret_cast<const void*>(lmm_read),
                            hipFuncAttributeMaxDynamicSharedMemorySize, 114688);

  lmm_prep<<<1, 256, 0, stream>>>(ln_g, ln_b, Wew, scal);
  lmm_wcat<<<(HD * NCOL) / 256, 256, 0, stream>>>(spw, wqkv, Wcat);
  lmm_lnstats<<<NT / 8, 256, 0, stream>>>(H, ln_g, Wew, mu, rstd, s3);
  lmm_gemm<<<NT / 64, 256, 0, stream>>>(H, mu, rstd, ln_g, ln_b, Wcat, spb, P, Qbf);
  lmm_segment<<<NB * NSEG, 128, 0, stream>>>(P, sumq, mu, rstd, s3, scal, Web, Zseg, etas);
  lmm_update<<<NB, 256, 65536, stream>>>(Zseg, etas, M, etach, temp, Mnew, Mbf, Mtbf, scal);
  lmm_finalize<<<1, 1, 0, stream>>>(scal, out);
  lmm_read<<<NT / LSEG, 256, 114688, stream>>>(Qbf, Mbf, Mtbf, out);
}
